// POLAR_84301618086507
// MI455X (gfx1250) — compile-verified, matrix operand order checked by reading, not run on hardware
//
#include <hip/hip_runtime.h>
#include <stdint.h>

typedef __attribute__((ext_vector_type(16))) _Float16 v16h;
typedef __attribute__((ext_vector_type(8)))  _Float16 v8h;
typedef __attribute__((ext_vector_type(8)))  float    v8f;

#define B_      2048
#define L_      128
#define NROWS   (B_*L_)
#define KDIM    4
#define ENC     108
#define ENCP    112      // 7*16
#define KP1     128      // layer2 K padded 108->128
#define D2      172
#define D3      512
#define D4      172
#define D4P     176      // 11*16
#define CIN     280      // ENC + D2
#define CINP    288      // 18*16
#define TM      32       // rows per workgroup
#define SRN     (2*TM)   // sub-rows (src/tgt sides)
#define NTHREADS 256
#define NWAVES  8
#define NP_FC1  (CINP/32)   // 9 K-panels
#define NP_FC2  (D3/32)     // 16 K-panels

// ---- workspace layout (f16 elements): panel-contiguous pre-converted weights
#define WS_P1    0                       // [9][512][32]  f16  147456 elts (32768 B/panel)
#define WS_P2    147456                  // [16][176][32] f16   90112 elts (11264 B/panel)
#define WS_W2T   237568                  // [112][128]    f16   14336 elts
#define WS_TOTAL 251904                  // 503808 bytes

#define P1_PANEL_BYTES 32768
#define P2_PANEL_BYTES 11264

// ---- LDS overlay (bytes) ----
#define OFF_X     0                      // f16 [TM][CINP]       18432
#define OFF_H1    18432                  // f16 [SRN][KP1]       16384  (ph1-2)
#define OFF_W2T   34816                  // f16 [ENCP][KP1]      28672  (ph0-2, end 63488)
#define OFF_BPAN  18432                  // f16 2x[D3][32]       65536  (ph3, end 83968)
#define OFF_H3    83968                  // f16 [TM][D3]         32768  (end 116736)
#define OFF_B2P   0                      // f16 2x[D4P][32]      22528  (ph4)
#define OFF_W1    116736                 // f32 [4][ENCP]         1792
#define OFF_B1    118528                 // f32 [ENCP]             448
#define OFF_B2    118976                 // f32 [ENCP]             448
#define OFF_F1B   119424                 // f32 [D3]              2048
#define OFF_F2B   121472                 // f32 [D4P]              704
#define OFF_ENCS  122176                 // f32 [SRN][4]          1024
#define SMEM_BYTES 123200

// ---------- CDNA5 async global->LDS (ASYNCcnt-tracked) -----------------
__device__ __forceinline__ void async_b128(uint32_t lds_off,
                                           const _Float16* gbase, int gbyte) {
  asm volatile("global_load_async_to_lds_b128 %0, %1, %2"
               :: "v"(lds_off), "v"(gbyte), "s"(gbase) : "memory");
}
__device__ __forceinline__ void wait_async0() {
  asm volatile("s_wait_asynccnt 0x0" ::: "memory");
}

__device__ __forceinline__ v16h lds_frag(const _Float16* base, int strideHalves,
                                         int lr, int half) {
  // CDNA5 16-bit A/B fragment: e<8 -> K = half*8+e ; e>=8 -> K = 16+half*8+(e-8)
  const _Float16* p = base + lr * strideHalves + half * 8;
  v8h lo = *(const v8h*)(p);
  v8h hi = *(const v8h*)(p + 16);
  return __builtin_shufflevector(lo, hi, 0,1,2,3,4,5,6,7,8,9,10,11,12,13,14,15);
}

__device__ __forceinline__ v8f wmma16(v16h a, v16h b, v8f c) {
  return __builtin_amdgcn_wmma_f32_16x16x32_f16(false, a, false, b, (short)0, c,
                                                false, false);
}

// ======================================================================
// Prep: transpose + f32->f16 all weights into panel-contiguous workspace
// ======================================================================
__global__ __launch_bounds__(NTHREADS) void POLAR_prep_weights(
    const float* __restrict__ W2, const float* __restrict__ fc1_w,
    const float* __restrict__ fc2_w, _Float16* __restrict__ ws)
{
  const int NP1 = NP_FC1 * D3 * 32;        // 147456
  const int NP2 = NP_FC2 * D4P * 32;       // 90112
  int i = blockIdx.x * NTHREADS + threadIdx.x;
  if (i >= WS_TOTAL) return;
  if (i < NP1) {                           // fc1_w^T panels [kp][n][k]
    int kp = i >> 14;                      // 512*32 = 16384
    int rem = i & 16383;
    int n = rem >> 5, k = rem & 31;
    int kg = kp * 32 + k;
    float v = (kg < CIN) ? fc1_w[kg * D3 + n] : 0.f;
    ws[i] = (_Float16)v;
  } else if (i < NP1 + NP2) {              // fc2_w^T panels [kp][n][k]
    int j = i - NP1;
    int kp = j / (D4P * 32);
    int rem = j % (D4P * 32);
    int n = rem >> 5, k = rem & 31;
    float v = (n < D4) ? fc2_w[(kp * 32 + k) * D4 + n] : 0.f;
    ws[i] = (_Float16)v;
  } else {                                 // W2^T [f][e], padded
    int j = i - NP1 - NP2;
    int f = j >> 7, c = j & 127;
    float v = (f < ENC && c < ENC) ? W2[c * ENC + f] : 0.f;
    ws[i] = (_Float16)v;
  }
}

// ======================================================================
// Fused gather + 2-layer encoder + concat + fc1 + fc2
// ======================================================================
__global__ __launch_bounds__(NTHREADS) void POLAR_fused_pe_mlp_kernel(
    const float* __restrict__ emb, const int* __restrict__ idx,
    const float* __restrict__ W1,  const float* __restrict__ b1,
    const float* __restrict__ b2,  const float* __restrict__ x2,
    const float* __restrict__ fc1_b, const float* __restrict__ fc2_b,
    const _Float16* __restrict__ wsw, float* __restrict__ out)
{
  __shared__ __align__(64) unsigned char smem[SMEM_BYTES];
  _Float16* xs    = (_Float16*)(smem + OFF_X);
  _Float16* h1s   = (_Float16*)(smem + OFF_H1);
  _Float16* w2ts  = (_Float16*)(smem + OFF_W2T);
  _Float16* bpan  = (_Float16*)(smem + OFF_BPAN);
  _Float16* h3s   = (_Float16*)(smem + OFF_H3);
  _Float16* b2pan = (_Float16*)(smem + OFF_B2P);
  float*    w1s   = (float*)(smem + OFF_W1);
  float*    b1s   = (float*)(smem + OFF_B1);
  float*    b2s   = (float*)(smem + OFF_B2);
  float*    f1bs  = (float*)(smem + OFF_F1B);
  float*    f2bs  = (float*)(smem + OFF_F2B);
  float*    encs  = (float*)(smem + OFF_ENCS);

  const int tid  = threadIdx.x;
  const int wid  = tid >> 5;
  const int lane = tid & 31;
  const int lr   = lane & 15;
  const int half = lane >> 4;
  const int row0 = blockIdx.x * TM;

  // ---------------- Phase 0: async-stage W2^T ; load small consts --------
  {
    uint32_t l = (uint32_t)(uintptr_t)w2ts;
    const _Float16* g = wsw + WS_W2T;
    for (int c = tid; c < (ENCP * KP1 * 2) / 16; c += NTHREADS)   // 1792 chunks
      async_b128(l + c * 16, g, c * 16);
  }
  for (int e = tid; e < KDIM * ENCP; e += NTHREADS) {
    int k = e / ENCP, c = e % ENCP;
    w1s[e] = (c < ENC) ? W1[k * ENC + c] : 0.f;
  }
  for (int e = tid; e < ENCP; e += NTHREADS) {
    b1s[e] = (e < ENC) ? b1[e] : 0.f;
    b2s[e] = (e < ENC) ? b2[e] : 0.f;
  }
  for (int e = tid; e < D3; e += NTHREADS)  f1bs[e] = fc1_b[e];
  for (int e = tid; e < D4P; e += NTHREADS) f2bs[e] = (e < D4) ? fc2_b[e] : 0.f;
  for (int e = tid; e < SRN * KDIM; e += NTHREADS) {        // emb_table gather
    int sr = e >> 2, k = e & 3;
    int r = sr >> 1, s = sr & 1;
    int g = idx[row0 + r];
    encs[e] = emb[(long long)g * 8 + s * 4 + k];
  }
  wait_async0();
  __syncthreads();

  // ---------------- Phase 1: layer1 (K=4, VALU) -> h1 [SRN][KP1] f16 ------
  for (int e = tid; e < SRN * KP1; e += NTHREADS) {
    int sr = e >> 7, c = e & 127;
    float v = 0.f;
    if (c < ENC) {
      float a = b1s[c];
      a += encs[sr * 4 + 0] * w1s[0 * ENCP + c];
      a += encs[sr * 4 + 1] * w1s[1 * ENCP + c];
      a += encs[sr * 4 + 2] * w1s[2 * ENCP + c];
      a += encs[sr * 4 + 3] * w1s[3 * ENCP + c];
      v = fmaxf(a, 0.f);
    }
    h1s[e] = (_Float16)v;
  }
  __syncthreads();

  // ---------------- Phase 2: layer2 WMMA [64x128]x[128x112], pair-sum -> pe
  for (int t = wid; t < 4 * 7; t += NWAVES) {               // wave-uniform
    int mt = t / 7, nt = t % 7;
    v8f acc = {};
#pragma unroll
    for (int kk = 0; kk < KP1 / 32; ++kk) {
      v16h a = lds_frag(h1s  + (mt * 16) * KP1 + kk * 32, KP1, lr, half);
      v16h b = lds_frag(w2ts + (nt * 16) * KP1 + kk * 32, KP1, lr, half);
      acc = wmma16(a, b, acc);
    }
    int col = nt * 16 + lr;
    if (col < ENC) {
      // D rows: acc[j] -> srow = mt*16 + half*8 + j ; sum (2p,2p+1) side pairs
#pragma unroll
      for (int j = 0; j < 4; ++j) {
        int p = mt * 8 + half * 4 + j;                      // pe row 0..31
        float v = acc[2 * j] + acc[2 * j + 1] + 2.f * b2s[col];
        xs[p * CINP + col] = (_Float16)v;
      }
    }
  }
  // Phase 2b: x2 -> concat columns [ENC..CIN), zero pad [CIN..CINP)
  for (int e = tid; e < TM * (CINP - ENC); e += NTHREADS) {
    int r = e / (CINP - ENC);
    int c = ENC + e % (CINP - ENC);
    float v = (c < CIN) ? x2[(long long)(row0 + r) * D2 + (c - ENC)] : 0.f;
    xs[r * CINP + c] = (_Float16)v;
  }
  __syncthreads();

  // ---------------- Phase 3: fc1  [32x288]x[288x512] -> relu -> h3 --------
  // double-buffered async panel streaming from workspace
  v8f acc1[8];
  v8f zero = {};
#pragma unroll
  for (int i = 0; i < 8; ++i) acc1[i] = zero;
  const _Float16* p1g = wsw + WS_P1;
  const uint32_t bpan_lds = (uint32_t)(uintptr_t)bpan;
  for (int c = tid; c < P1_PANEL_BYTES / 16; c += NTHREADS)   // panel 0 -> buf 0
    async_b128(bpan_lds + c * 16, p1g, c * 16);
  wait_async0();
  __syncthreads();
  for (int kp = 0; kp < NP_FC1; ++kp) {
    if (kp + 1 < NP_FC1) {                                  // prefetch panel kp+1
      uint32_t ln = bpan_lds + ((kp + 1) & 1) * P1_PANEL_BYTES;
      int gb = (kp + 1) * P1_PANEL_BYTES;
      for (int c = tid; c < P1_PANEL_BYTES / 16; c += NTHREADS)
        async_b128(ln + c * 16, p1g, gb + c * 16);
    }
    const _Float16* bcur = bpan + (kp & 1) * (P1_PANEL_BYTES / 2);
#pragma unroll
    for (int i = 0; i < 8; ++i) {
      int t = wid * 8 + i;
      int mt = t >> 5, nt = t & 31;
      v16h a = lds_frag(xs + (mt * 16) * CINP + kp * 32, CINP, lr, half);
      v16h b = lds_frag(bcur + (nt * 16) * 32, 32, lr, half);
      acc1[i] = wmma16(a, b, acc1[i]);
    }
    wait_async0();
    __syncthreads();
  }
#pragma unroll
  for (int i = 0; i < 8; ++i) {                             // epilogue -> h3 f16
    int t = wid * 8 + i;
    int mt = t >> 5, nt = t & 31;
    int col = nt * 16 + lr;
#pragma unroll
    for (int j = 0; j < 8; ++j) {
      int r = mt * 16 + half * 8 + j;
      float v = fmaxf(acc1[i][j] + f1bs[col], 0.f);
      h3s[r * D3 + col] = (_Float16)v;
    }
  }
  __syncthreads();

  // ---------------- Phase 4: fc2  [32x512]x[512x176] -> out ---------------
  v8f acc2[3];
#pragma unroll
  for (int j = 0; j < 3; ++j) acc2[j] = zero;
  const _Float16* p2g = wsw + WS_P2;
  const uint32_t b2p_lds = (uint32_t)(uintptr_t)b2pan;
  for (int c = tid; c < P2_PANEL_BYTES / 16; c += NTHREADS)   // panel 0 -> buf 0
    async_b128(b2p_lds + c * 16, p2g, c * 16);
  wait_async0();
  __syncthreads();
  for (int kp = 0; kp < NP_FC2; ++kp) {
    if (kp + 1 < NP_FC2) {
      uint32_t ln = b2p_lds + ((kp + 1) & 1) * P2_PANEL_BYTES;
      int gb = (kp + 1) * P2_PANEL_BYTES;
      for (int c = tid; c < P2_PANEL_BYTES / 16; c += NTHREADS)
        async_b128(ln + c * 16, p2g, gb + c * 16);
    }
    const _Float16* bcur = b2pan + (kp & 1) * (P2_PANEL_BYTES / 2);
#pragma unroll
    for (int j = 0; j < 3; ++j) {
      int t = wid + NWAVES * j;                             // wave-uniform branch
      if (t < 2 * 11) {
        int mt = t / 11, nt = t % 11;
        v16h a = lds_frag(h3s + (mt * 16) * D3 + kp * 32, D3, lr, half);
        v16h b = lds_frag(bcur + (nt * 16) * 32, 32, lr, half);
        acc2[j] = wmma16(a, b, acc2[j]);
      }
    }
    wait_async0();
    __syncthreads();
  }
#pragma unroll
  for (int j = 0; j < 3; ++j) {                             // bias + store f32
    int t = wid + NWAVES * j;
    if (t < 22) {
      int mt = t / 11, nt = t % 11;
      int col = nt * 16 + lr;
      if (col < D4) {
#pragma unroll
        for (int jr = 0; jr < 8; ++jr) {
          int gr = row0 + mt * 16 + half * 8 + jr;
          out[(long long)gr * D4 + col] = acc2[j][jr] + f2bs[col];
        }
      }
    }
  }
}

extern "C" void kernel_launch(void* const* d_in, const int* in_sizes, int n_in,
                              void* d_out, int out_size, void* d_ws, size_t ws_size,
                              hipStream_t stream) {
  const float* emb  = (const float*)d_in[0];
  const int*   idx  = (const int*)d_in[1];
  const float* W1   = (const float*)d_in[2];
  const float* b1   = (const float*)d_in[3];
  const float* W2   = (const float*)d_in[4];
  const float* b2   = (const float*)d_in[5];
  const float* x2   = (const float*)d_in[6];
  const float* f1w  = (const float*)d_in[7];
  const float* f1b  = (const float*)d_in[8];
  const float* f2w  = (const float*)d_in[9];
  const float* f2b  = (const float*)d_in[10];
  float* outp = (float*)d_out;
  _Float16* wsw = (_Float16*)d_ws;
  (void)in_sizes; (void)n_in; (void)out_size; (void)ws_size;

  dim3 block(NTHREADS);
  dim3 gridPrep((WS_TOTAL + NTHREADS - 1) / NTHREADS);
  hipLaunchKernelGGL(POLAR_prep_weights, gridPrep, block, 0, stream,
                     W2, f1w, f2w, wsw);

  dim3 gridMain(NROWS / TM);
  hipLaunchKernelGGL(POLAR_fused_pe_mlp_kernel, gridMain, block, 0, stream,
                     emb, idx, W1, b1, b2, x2, f1b, f2b,
                     (const _Float16*)wsw, outp);
}
